// MultiHeadAttention_88622355185721
// MI455X (gfx1250) — matrix operand order repaired and checked by reading, not compiled, not run on hardware
//
#include <hip/hip_runtime.h>
#include <hip/hip_bf16.h>

#define D_ATTN 1024
#define NHEADS 16
#define DK     64
#define BATCH  4
#define SEQ    2048
#define BS     (BATCH * SEQ)
#define SROW   2056   // attention LDS row stride (floats): 2056 % 64 = 8 -> banks spread
#define ALDA   1032   // GEMM LDS A-tile row stride (bf16): 2064B/4 = 516 words, 516 % 64 = 4

typedef __bf16 bf16;
typedef __attribute__((ext_vector_type(16))) __bf16 v16bf;
typedef __attribute__((ext_vector_type(8)))  __bf16 bf16x8;
typedef __attribute__((ext_vector_type(8)))  float  v8f;
typedef __attribute__((ext_vector_type(4)))  float  f32x4;
typedef __attribute__((ext_vector_type(4)))  int    v4i;

#if defined(__has_builtin)
#  if __has_builtin(__builtin_amdgcn_global_load_async_to_lds_b128)
#    define HAVE_ASYNC_B128 1
#  endif
#  if __has_builtin(__builtin_amdgcn_s_wait_asynccnt)
#    define HAVE_WAIT_ASYNC 1
#  endif
#endif

// ---- CDNA5 async global->LDS copy (ASYNCcnt-tracked) ---------------------
__device__ __forceinline__ void async_ld16(const bf16* g, bf16* l) {
#ifdef HAVE_ASYNC_B128
    __builtin_amdgcn_global_load_async_to_lds_b128(
        (__attribute__((address_space(1))) v4i*)g,
        (__attribute__((address_space(3))) v4i*)l, 0, 0);
#else
    asm volatile("global_load_async_to_lds_b128 %0, %1, off"
                 :: "v"((unsigned int)(size_t)l), "v"(g)
                 : "memory");
#endif
}

__device__ __forceinline__ void wait_async0() {
#ifdef HAVE_WAIT_ASYNC
    __builtin_amdgcn_s_wait_asynccnt(0);
#else
    asm volatile("s_wait_asynccnt 0x0" ::: "memory");
#endif
}

// ---- WMMA helpers --------------------------------------------------------
__device__ __forceinline__ v8f wmma_bf16(v16bf a, v16bf b, v8f c) {
    return __builtin_amdgcn_wmma_f32_16x16x32_bf16(
        false, a, false, b, (short)0, c, false, false);
}

__device__ __forceinline__ v8f zero_v8f() {
    v8f z;
#pragma unroll
    for (int i = 0; i < 8; ++i) z[i] = 0.0f;
    return z;
}

// A fragment: 16x32 bf16, src row-major [m][k], lda elements (global or LDS).
__device__ __forceinline__ v16bf load_a16(const bf16* base, int lda, int lane) {
    const int m = lane & 15, h = lane >> 4;
    const bf16* p0 = base + (size_t)m * lda + 8 * h;
    const bf16* p1 = base + (size_t)m * lda + 16 + 8 * h;
    bf16x8 lo = *(const bf16x8*)p0;
    bf16x8 hi = *(const bf16x8*)p1;
    v16bf r;
#pragma unroll
    for (int i = 0; i < 8; ++i) { r[i] = lo[i]; r[8 + i] = hi[i]; }
    return r;
}

// B fragment: 32x16 bf16 (KxN) supplied as B^T row-major: base[n*ldb + k].
__device__ __forceinline__ v16bf load_b16(const bf16* base, int ldb, int lane) {
    const int n = lane & 15, h = lane >> 4;
    const bf16* p = base + (size_t)n * ldb + 8 * h;
    bf16x8 lo = *(const bf16x8*)p;
    bf16x8 hi = *(const bf16x8*)(p + 16);
    v16bf r;
#pragma unroll
    for (int i = 0; i < 8; ++i) { r[i] = lo[i]; r[8 + i] = hi[i]; }
    return r;
}

// ---- precision conversion kernels ---------------------------------------
__global__ void cvt_x_kernel(const float* __restrict__ x, bf16* __restrict__ xb, int n) {
    int i = blockIdx.x * 256 + threadIdx.x;
    if (i < n) xb[i] = (bf16)x[i];
}

__global__ void cvt_wT_kernel(const float* __restrict__ w0, const float* __restrict__ w1,
                              const float* __restrict__ w2, const float* __restrict__ w3,
                              bf16* __restrict__ o0, bf16* __restrict__ o1,
                              bf16* __restrict__ o2, bf16* __restrict__ o3) {
    const float* w = (blockIdx.z == 0) ? w0 : (blockIdx.z == 1) ? w1 : (blockIdx.z == 2) ? w2 : w3;
    bf16* o = (blockIdx.z == 0) ? o0 : (blockIdx.z == 1) ? o1 : (blockIdx.z == 2) ? o2 : o3;
    int i = blockIdx.x * 256 + threadIdx.x;
    int n = i >> 10;
    int k = i & (D_ATTN - 1);
    o[i] = (bf16)w[(size_t)k * D_ATTN + n];
}

// ---- QKV projection: y = x @ W + b --------------------------------------
// grid (4, 512, 3), block 128. Block: 16 rows x 256 cols. Wave: 16x64 (4 WMMA tiles).
// A tile (16x1024) async-staged to LDS once, shared by all 4 waves.
__global__ void __launch_bounds__(128)
qkv_gemm_kernel(const bf16* __restrict__ xb,
                const bf16* __restrict__ wqT, const bf16* __restrict__ wkT,
                const bf16* __restrict__ wvT,
                const float* __restrict__ bq, const float* __restrict__ bk,
                const float* __restrict__ bv,
                bf16* __restrict__ q_b, bf16* __restrict__ k_b, bf16* __restrict__ vT_b,
                float* __restrict__ k_out, float* __restrict__ v_out) {
    __shared__ bf16 aTile[16 * ALDA];               // ~33 KB
    const int tid  = threadIdx.x;
    const int lane = tid & 31;
    const int wid  = tid >> 5;
    const int rt = blockIdx.y;
    const int which = blockIdx.z;

    const bf16* wT   = (which == 0) ? wqT : (which == 1) ? wkT : wvT;
    const float* bia = (which == 0) ? bq  : (which == 1) ? bk  : bv;

    // async-stage A tile: 16 rows x 1024 bf16
    const bf16* aSrc = xb + (size_t)rt * 16 * D_ATTN;
    for (int c = tid; c < 16 * (D_ATTN / 8); c += 128) {
        int row = c >> 7, off = (c & 127) * 8;
        async_ld16(aSrc + (size_t)row * D_ATTN + off, aTile + row * ALDA + off);
    }
    wait_async0();
    __syncthreads();

    const int ct0 = blockIdx.x * 16 + wid * 4;      // first of 4 col tiles for this wave
    const bf16* brow = wT + (size_t)ct0 * 16 * D_ATTN;

    v8f acc[4];
#pragma unroll
    for (int j = 0; j < 4; ++j) acc[j] = zero_v8f();

#pragma unroll 2
    for (int kk = 0; kk < D_ATTN; kk += 32) {
        v16bf a = load_a16(aTile + kk, ALDA, lane);
#pragma unroll
        for (int j = 0; j < 4; ++j) {
            const bf16* bj = brow + (size_t)j * 16 * D_ATTN;
            __builtin_prefetch(bj + kk + 256, 0, 3);    // global_prefetch_b8
            v16bf b = load_b16(bj + kk, D_ATTN, lane);
            acc[j] = wmma_bf16(a, b, acc[j]);
        }
    }

    const int n = lane & 15, half = lane >> 4;
#pragma unroll
    for (int j = 0; j < 4; ++j) {
        const int dcol = (ct0 + j) * 16 + n;
        const float bval = bia[dcol];
        const int hh = dcol >> 6, dk = dcol & 63;
#pragma unroll
        for (int r = 0; r < 8; ++r) {
            int m   = r + 8 * half;
            int row = rt * 16 + m;
            int bi  = row >> 11, s = row & (SEQ - 1);
            float v = acc[j][r] + bval;
            size_t hidx = (((size_t)bi * NHEADS + hh) * SEQ + s) * DK + dk;
            if (which == 0) {
                q_b[hidx] = (bf16)v;
            } else if (which == 1) {
                k_b[hidx]  = (bf16)v;
                k_out[hidx] = v;
            } else {
                v_out[hidx] = v;
                vT_b[(((size_t)bi * NHEADS + hh) * DK + dk) * SEQ + s] = (bf16)v;
            }
        }
    }
}

// ---- fused attention: scores(WMMA) -> softmax(LDS) -> P@V(WMMA) ----------
// grid (128, 16, 4), block 256 (8 waves), dyn LDS = 16*SROW*4 bytes
__global__ void attention_kernel(const bf16* __restrict__ q_b, const bf16* __restrict__ k_b,
                                 const bf16* __restrict__ vT_b,
                                 float* __restrict__ attn_out, bf16* __restrict__ ctx_b) {
    extern __shared__ float sc[];              // [16][SROW]
    const int lane = threadIdx.x & 31;
    const int wid  = threadIdx.x >> 5;
    const int qt = blockIdx.x;
    const int hh = blockIdx.y;
    const int bi = blockIdx.z;
    const size_t head_off = ((size_t)bi * NHEADS + hh) * SEQ;
    const int n = lane & 15, half = lane >> 4;

    // --- phase 1: S = (q @ k^T)/8, causal mask, into LDS ---
    const bf16* qbase = q_b + (head_off + (size_t)qt * 16) * DK;
    v16bf aq0 = load_a16(qbase, DK, lane);
    v16bf aq1 = load_a16(qbase + 32, DK, lane);
    const int ntiles = qt + 1;
    for (int kt = wid; kt < ntiles; kt += 8) {
        if (kt + 8 < ntiles)                   // prefetch next K tile (2 KB)
            __builtin_prefetch(k_b + (head_off + (size_t)(kt + 8) * 16) * DK + lane * 32, 0, 3);
        const bf16* kbase = k_b + (head_off + (size_t)kt * 16) * DK;
        v16bf b0 = load_b16(kbase, DK, lane);
        v16bf b1 = load_b16(kbase + 32, DK, lane);
        v8f acc = zero_v8f();
        acc = wmma_bf16(aq0, b0, acc);
        acc = wmma_bf16(aq1, b1, acc);
#pragma unroll
        for (int r = 0; r < 8; ++r) {
            int m  = r + 8 * half;
            int qi = qt * 16 + m;
            int ki = kt * 16 + n;
            float s = acc[r] * 0.125f;
            if (ki > qi) s = -3.0e38f;
            sc[m * SROW + ki] = s;
        }
    }
    __syncthreads();

    // --- phase 2: row softmax; attn f32 out; probs back into LDS ---
    const int L   = ntiles * 16;
    const int L32 = (L + 31) & ~31;
    for (int rr = 0; rr < 2; ++rr) {
        int m  = wid * 2 + rr;
        int qi = qt * 16 + m;
        float* row = sc + m * SROW;
        float mx = -3.0e38f;
        for (int c = lane; c < L; c += 32) mx = fmaxf(mx, row[c]);
#pragma unroll
        for (int off = 16; off >= 1; off >>= 1) mx = fmaxf(mx, __shfl_xor(mx, off, 32));
        float sum = 0.0f;
        for (int c = lane; c < L; c += 32) {
            float p = __expf(row[c] - mx);
            row[c] = p;
            sum += p;
        }
#pragma unroll
        for (int off = 16; off >= 1; off >>= 1) sum += __shfl_xor(sum, off, 32);
        float inv = 1.0f / sum;
        float* arow_out = attn_out + (head_off + qi) * SEQ;
        for (int c = lane; c < L; c += 32) {
            float p = row[c] * inv;
            row[c] = p;
            arow_out[c] = p;
        }
        for (int c = L + lane; c < L32; c += 32) row[c] = 0.0f;
        for (int c = L + lane; c < SEQ; c += 32) arow_out[c] = 0.0f;
    }
    __syncthreads();

    // --- phase 3: ctx = P @ V ; waves 0..3 own one Dk col tile each ---
    if (wid < 4) {
        const int ct = wid;
        const bf16* vbase = vT_b + (((size_t)bi * NHEADS + hh) * DK + ct * 16) * SEQ;
        v8f acc = zero_v8f();
        const int m = lane & 15;
        for (int k0 = 0; k0 < L32; k0 += 32) {
            __builtin_prefetch(vbase + (size_t)n * SEQ + k0 + 64, 0, 3);
            const float* pr = sc + m * SROW + k0;
            f32x4 c0 = *(const f32x4*)(pr + 8 * half);
            f32x4 c1 = *(const f32x4*)(pr + 8 * half + 4);
            f32x4 c2 = *(const f32x4*)(pr + 16 + 8 * half);
            f32x4 c3 = *(const f32x4*)(pr + 16 + 8 * half + 4);
            v16bf a;
#pragma unroll
            for (int i = 0; i < 4; ++i) {
                a[i]      = (bf16)c0[i];
                a[4 + i]  = (bf16)c1[i];
                a[8 + i]  = (bf16)c2[i];
                a[12 + i] = (bf16)c3[i];
            }
            const bf16* p = vbase + (size_t)n * SEQ + k0 + 8 * half;
            bf16x8 lo = *(const bf16x8*)p;
            bf16x8 hi = *(const bf16x8*)(p + 16);
            v16bf b;
#pragma unroll
            for (int i = 0; i < 8; ++i) { b[i] = lo[i]; b[8 + i] = hi[i]; }
            acc = wmma_bf16(a, b, acc);
        }
#pragma unroll
        for (int r = 0; r < 8; ++r) {
            int mm = r + 8 * half;
            int qi = qt * 16 + mm;
            size_t o = ((size_t)bi * SEQ + qi) * D_ATTN + hh * DK + ct * 16 + n;
            ctx_b[o] = (bf16)acc[r];
        }
    }
}

// ---- output projection: out = ctx @ Wo + bo (f32) ------------------------
// grid (4, 512), block 128; same blocking as qkv_gemm
__global__ void __launch_bounds__(128)
out_proj_kernel(const bf16* __restrict__ ctx_b, const bf16* __restrict__ woT,
                const float* __restrict__ bo, float* __restrict__ out) {
    __shared__ bf16 aTile[16 * ALDA];
    const int tid  = threadIdx.x;
    const int lane = tid & 31;
    const int wid  = tid >> 5;
    const int rt = blockIdx.y;

    const bf16* aSrc = ctx_b + (size_t)rt * 16 * D_ATTN;
    for (int c = tid; c < 16 * (D_ATTN / 8); c += 128) {
        int row = c >> 7, off = (c & 127) * 8;
        async_ld16(aSrc + (size_t)row * D_ATTN + off, aTile + row * ALDA + off);
    }
    wait_async0();
    __syncthreads();

    const int ct0 = blockIdx.x * 16 + wid * 4;
    const bf16* brow = woT + (size_t)ct0 * 16 * D_ATTN;

    v8f acc[4];
#pragma unroll
    for (int j = 0; j < 4; ++j) acc[j] = zero_v8f();

#pragma unroll 2
    for (int kk = 0; kk < D_ATTN; kk += 32) {
        v16bf a = load_a16(aTile + kk, ALDA, lane);
#pragma unroll
        for (int j = 0; j < 4; ++j) {
            const bf16* bj = brow + (size_t)j * 16 * D_ATTN;
            __builtin_prefetch(bj + kk + 256, 0, 3);
            v16bf b = load_b16(bj + kk, D_ATTN, lane);
            acc[j] = wmma_bf16(a, b, acc[j]);
        }
    }

    const int n = lane & 15, half = lane >> 4;
#pragma unroll
    for (int j = 0; j < 4; ++j) {
        const float bval = bo[(ct0 + j) * 16 + n];
#pragma unroll
        for (int r = 0; r < 8; ++r) {
            int m = r + 8 * half;
            out[((size_t)(rt * 16 + m)) * D_ATTN + (ct0 + j) * 16 + n] = acc[j][r] + bval;
        }
    }
}

// ---- host side -----------------------------------------------------------
extern "C" void kernel_launch(void* const* d_in, const int* in_sizes, int n_in,
                              void* d_out, int out_size, void* d_ws, size_t ws_size,
                              hipStream_t stream) {
    const float* x  = (const float*)d_in[0];
    const float* Wq = (const float*)d_in[1];
    const float* bq = (const float*)d_in[2];
    const float* Wk = (const float*)d_in[3];
    const float* bk = (const float*)d_in[4];
    const float* Wv = (const float*)d_in[5];
    const float* bv = (const float*)d_in[6];
    const float* Wo = (const float*)d_in[7];
    const float* bo = (const float*)d_in[8];

    char* ws = (char*)d_ws;
    size_t off = 0;
    auto alloc = [&](size_t bytes) -> void* {
        void* p = ws + off;
        off += (bytes + 255) & ~(size_t)255;
        return p;
    };
    bf16* xb    = (bf16*)alloc((size_t)BS * D_ATTN * sizeof(bf16));
    bf16* wqT   = (bf16*)alloc((size_t)D_ATTN * D_ATTN * sizeof(bf16));
    bf16* wkT   = (bf16*)alloc((size_t)D_ATTN * D_ATTN * sizeof(bf16));
    bf16* wvT   = (bf16*)alloc((size_t)D_ATTN * D_ATTN * sizeof(bf16));
    bf16* woT   = (bf16*)alloc((size_t)D_ATTN * D_ATTN * sizeof(bf16));
    bf16* q_b   = (bf16*)alloc((size_t)BS * D_ATTN * sizeof(bf16));   // [B,H,S,Dk]
    bf16* k_b   = (bf16*)alloc((size_t)BS * D_ATTN * sizeof(bf16));   // [B,H,S,Dk]
    bf16* vT_b  = (bf16*)alloc((size_t)BS * D_ATTN * sizeof(bf16));   // [B,H,Dk,S]
    bf16* ctx_b = (bf16*)alloc((size_t)BS * D_ATTN * sizeof(bf16));   // [B,S,D]

    float* out  = (float*)d_out;
    float* attn = out + (size_t)BS * D_ATTN;
    float* kout = attn + (size_t)BATCH * NHEADS * SEQ * SEQ;
    float* vout = kout + (size_t)BS * D_ATTN;

    cvt_x_kernel<<<dim3((BS * D_ATTN + 255) / 256), 256, 0, stream>>>(x, xb, BS * D_ATTN);
    cvt_wT_kernel<<<dim3(D_ATTN * D_ATTN / 256, 1, 4), 256, 0, stream>>>(
        Wq, Wk, Wv, Wo, wqT, wkT, wvT, woT);
    qkv_gemm_kernel<<<dim3(4, 512, 3), 128, 0, stream>>>(
        xb, wqT, wkT, wvT, bq, bk, bv, q_b, k_b, vT_b, kout, vout);
    attention_kernel<<<dim3(SEQ / 16, NHEADS, BATCH), 256, 16 * SROW * sizeof(float), stream>>>(
        q_b, k_b, vT_b, attn, ctx_b);
    out_proj_kernel<<<dim3(4, 512), 128, 0, stream>>>(ctx_b, woT, bo, out);
}
